// block_15341623181397
// MI455X (gfx1250) — matrix operand order repaired and checked by reading, not compiled, not run on hardware
//
#include <hip/hip_runtime.h>
#include <hip/hip_bf16.h>
#include <math.h>

typedef _Float16 half_t;
typedef __attribute__((ext_vector_type(16))) _Float16 v16h;
typedef __attribute__((ext_vector_type(8)))  float    v8f;

#define TB   2
#define TS   2048
#define TE   768
#define TH   12
#define TD   64
#define TM   (TB * TS)        // 4096 rows
#define T3E  (3 * TE)
#define T4E  (4 * TE)

// ---------------------------------------------------------------------------
// gfx1250 async global->LDS copy path (ASYNCcnt-tracked)
// ---------------------------------------------------------------------------
#if defined(__gfx1250__) && __has_builtin(__builtin_amdgcn_global_load_async_to_lds_b128)
#define HAVE_ASYNC 1
typedef int v4i __attribute__((vector_size(16)));
typedef __attribute__((address_space(1))) v4i* gvp;
typedef __attribute__((address_space(3))) v4i* lvp;
#else
#define HAVE_ASYNC 0
#endif

// copy 16 halves (32 bytes) global -> LDS, per calling thread
__device__ __forceinline__ void stage32B(half_t* __restrict__ dst,
                                         const half_t* __restrict__ src) {
#if HAVE_ASYNC
  __builtin_amdgcn_global_load_async_to_lds_b128((gvp)src,       (lvp)dst,       0, 0);
  __builtin_amdgcn_global_load_async_to_lds_b128((gvp)(src + 8), (lvp)(dst + 8), 0, 0);
#else
  ((uint4*)dst)[0] = ((const uint4*)src)[0];
  ((uint4*)dst)[1] = ((const uint4*)src)[1];
#endif
}

__device__ __forceinline__ void wait_async() {
#if HAVE_ASYNC
#if __has_builtin(__builtin_amdgcn_s_wait_asynccnt)
  __builtin_amdgcn_s_wait_asynccnt(0);
#else
  asm volatile("s_wait_asynccnt 0x0" ::: "memory");
#endif
#endif
}

// ---------------------------------------------------------------------------
// Flat f32 -> f16 cast (weights)
// ---------------------------------------------------------------------------
__global__ __launch_bounds__(256) void cast_kernel(const float* __restrict__ x,
                                                   half_t* __restrict__ y) {
  const int i = blockIdx.x * 256 + threadIdx.x;
  y[i] = (half_t)x[i];
}

// ---------------------------------------------------------------------------
// LayerNorm: one 256-thread block per row of E=768, f32 in, f16 out
// ---------------------------------------------------------------------------
__global__ __launch_bounds__(256) void ln_kernel(const float* __restrict__ x,
                                                 const float* __restrict__ g,
                                                 const float* __restrict__ b,
                                                 half_t* __restrict__ y) {
  const int row = blockIdx.x;
  const int t = threadIdx.x;
  const float* xr = x + (size_t)row * TE;
  float v0 = xr[t], v1 = xr[t + 256], v2 = xr[t + 512];
  __shared__ float ssum[256];
  __shared__ float ssq[256];
  ssum[t] = v0 + v1 + v2;
  ssq[t]  = v0 * v0 + v1 * v1 + v2 * v2;
  __syncthreads();
  for (int off = 128; off > 0; off >>= 1) {
    if (t < off) { ssum[t] += ssum[t + off]; ssq[t] += ssq[t + off]; }
    __syncthreads();
  }
  const float mean = ssum[0] * (1.0f / TE);
  const float var  = ssq[0] * (1.0f / TE) - mean * mean;
  const float inv  = rsqrtf(var + 1e-5f);
  half_t* yr = y + (size_t)row * TE;
  yr[t]       = (half_t)((v0 - mean) * inv * g[t]       + b[t]);
  yr[t + 256] = (half_t)((v1 - mean) * inv * g[t + 256] + b[t + 256]);
  yr[t + 512] = (half_t)((v2 - mean) * inv * g[t + 512] + b[t + 512]);
}

// ---------------------------------------------------------------------------
// GEMM: act(A[M,K] @ W[N,K]^T + bias) [+ residual] -> f32 and/or f16 outputs
// 256 threads = 8 waves; block tile 128x128; wave tile 32x64; K step 32.
// Double-buffered LDS with async global->LDS prefetch of tile k+1 overlapped
// with WMMA on tile k. f16 operands, f32 accumulate. act: 0 none, 1 GELU.
// ---------------------------------------------------------------------------
#define LDSS 40  // padded half stride per 32-wide K slice

__global__ __launch_bounds__(256) void gemm_wmma(const half_t* __restrict__ A,
                                                 const half_t* __restrict__ W,
                                                 const float* __restrict__ bias,
                                                 const float* __restrict__ residual,
                                                 float* __restrict__ Cf,
                                                 half_t* __restrict__ Ch,
                                                 int M, int N, int K, int act) {
  __shared__ half_t As[2][128 * LDSS];
  __shared__ half_t Ws[2][128 * LDSS];

  const int tid  = threadIdx.x;
  const int wave = tid >> 5;
  const int lane = tid & 31;
  const int row  = lane & 15;
  const int hi   = lane >> 4;
  const int wm   = (wave & 3) * 32;   // wave row offset in block tile
  const int wn   = (wave >> 2) * 64;  // wave col offset in block tile

  const int ntile = N >> 7;
  const int tm = (blockIdx.x / ntile) * 128;
  const int tn = (blockIdx.x % ntile) * 128;

  v8f acc[2][4];
#pragma unroll
  for (int s = 0; s < 2; ++s)
#pragma unroll
    for (int t = 0; t < 4; ++t)
#pragma unroll
      for (int v = 0; v < 8; ++v) acc[s][t][v] = 0.0f;

  const int lr = tid >> 1;         // 0..127: tile row staged by this thread
  const int lc = (tid & 1) * 16;   // half-row of 16 halves

  const half_t* srcA = A + (size_t)(tm + lr) * K + lc;
  const half_t* srcW = W + (size_t)(tn + lr) * K + lc;
  const int lofs = lr * LDSS + lc;
  const int NT = K >> 5;

  // Prologue: stage k-tile 0 into buffer 0
  stage32B(&As[0][lofs], srcA);
  stage32B(&Ws[0][lofs], srcW);

  for (int it = 0; it < NT; ++it) {
    const int cur = it & 1;
    wait_async();        // my async writes into buf[cur] done
    __syncthreads();     // all waves' writes done; all prior reads of buf[cur^1] done

    // Prefetch next k-tile into the other buffer, overlapped with compute
    if (it + 1 < NT) {
      const int k0n = (it + 1) << 5;
      stage32B(&As[cur ^ 1][lofs], srcA + k0n);
      stage32B(&Ws[cur ^ 1][lofs], srcW + k0n);
    }

    // Two A fragments (rows wm..wm+15, wm+16..wm+31); hi half K+16
    v16h af[2];
#pragma unroll
    for (int s = 0; s < 2; ++s) {
      const half_t* ap = &As[cur][(wm + s * 16 + row) * LDSS + (hi ? 16 : 0)];
#pragma unroll
      for (int i = 0; i < 8; ++i) {
        const int kk = 2 * i;
        af[s][2 * i]     = ap[kk];
        af[s][2 * i + 1] = ap[kk + 1];
      }
    }
    // Four B fragments, each reused by both A fragments; hi half K+16
#pragma unroll
    for (int t = 0; t < 4; ++t) {
      v16h bf;
      const half_t* bp = &Ws[cur][(wn + t * 16 + row) * LDSS + (hi ? 16 : 0)];
#pragma unroll
      for (int j = 0; j < 8; ++j) {
        bf[2 * j]     = bp[2 * j];
        bf[2 * j + 1] = bp[2 * j + 1];
      }
      acc[0][t] = __builtin_amdgcn_wmma_f32_16x16x32_f16(false, af[0], false, bf,
                                                         (short)0, acc[0][t], false, false);
      acc[1][t] = __builtin_amdgcn_wmma_f32_16x16x32_f16(false, af[1], false, bf,
                                                         (short)0, acc[1][t], false, false);
    }
  }

  // Epilogue: C layout row = v + 8*hi, col = lane&15
#pragma unroll
  for (int s = 0; s < 2; ++s) {
    const int orow = tm + wm + s * 16 + hi * 8;
#pragma unroll
    for (int t = 0; t < 4; ++t) {
      const int ocol = tn + wn + t * 16 + row;
      const float bv = bias[ocol];
#pragma unroll
      for (int v = 0; v < 8; ++v) {
        const size_t r = (size_t)(orow + v);
        float val = acc[s][t][v] + bv;
        if (act == 1) val = 0.5f * val * (1.0f + erff(val * 0.70710678118f));
        if (residual) val += residual[r * N + ocol];
        if (Cf) Cf[r * N + ocol] = val;
        if (Ch) Ch[r * N + ocol] = (half_t)val;
      }
    }
  }
}

// ---------------------------------------------------------------------------
// Flash attention (no scale, no mask, full softmax), one wave per 16-query
// tile of one head. f16 qkv in, f16 out. 4 waves/block, uniform trip counts.
// ---------------------------------------------------------------------------
__global__ __launch_bounds__(128) void attn_kernel(const half_t* __restrict__ qkv,
                                                   half_t* __restrict__ out) {
  __shared__ half_t pbuf[4][16 * 32];

  const int wave = threadIdx.x >> 5;
  const int lane = threadIdx.x & 31;
  const int row  = lane & 15;
  const int hi   = lane >> 4;

  const int g   = blockIdx.x * 4 + wave;        // global query tile id
  const int b   = g / (TH * (TS / 16));
  const int rem = g % (TH * (TS / 16));
  const int h   = rem / (TS / 16);
  const int qt  = rem % (TS / 16);

  const size_t base = (size_t)b * TS * T3E;
  const half_t* Qp = qkv + base + (size_t)(qt * 16) * T3E + h * TD;
  const half_t* Kp = qkv + base + TE + h * TD;
  const half_t* Vp = qkv + base + 2 * TE + h * TD;

  // Q fragments for d=0..31 and d=32..63
  v16h qf[2];
#pragma unroll
  for (int f = 0; f < 2; ++f) {
    const half_t* qp = Qp + (size_t)row * T3E + f * 32 + (hi ? 16 : 0);
#pragma unroll
    for (int i = 0; i < 8; ++i) {
      const int kk = 2 * i;
      qf[f][2 * i]     = qp[kk];
      qf[f][2 * i + 1] = qp[kk + 1];
    }
  }

  v8f m, l, o[4];
#pragma unroll
  for (int v = 0; v < 8; ++v) { m[v] = -1e30f; l[v] = 0.0f; }
#pragma unroll
  for (int t = 0; t < 4; ++t)
#pragma unroll
    for (int v = 0; v < 8; ++v) o[t][v] = 0.0f;

  half_t* pw = pbuf[wave];

  for (int kc = 0; kc < TS; kc += 32) {
    // ---- scores S = Q K^T for 32 keys (two 16-key tiles) ----
    v8f sc[2];
#pragma unroll
    for (int t = 0; t < 2; ++t) {
#pragma unroll
      for (int v = 0; v < 8; ++v) sc[t][v] = 0.0f;
      const half_t* kp = Kp + (size_t)(kc + t * 16 + row) * T3E;
#pragma unroll
      for (int f = 0; f < 2; ++f) {
        v16h bf;
        const half_t* kpf = kp + f * 32 + (hi ? 16 : 0);
#pragma unroll
        for (int j = 0; j < 8; ++j) {
          bf[2 * j]     = kpf[2 * j];
          bf[2 * j + 1] = kpf[2 * j + 1];
        }
        sc[t] = __builtin_amdgcn_wmma_f32_16x16x32_f16(false, qf[f], false, bf,
                                                       (short)0, sc[t], false, false);
      }
    }

    // ---- online softmax: row max over 32 keys ----
    v8f cm;
#pragma unroll
    for (int v = 0; v < 8; ++v) cm[v] = fmaxf(sc[0][v], sc[1][v]);
#pragma unroll
    for (int off = 1; off < 16; off <<= 1)
#pragma unroll
      for (int v = 0; v < 8; ++v) cm[v] = fmaxf(cm[v], __shfl_xor(cm[v], off, 32));

    v8f mnew, alpha, rs;
#pragma unroll
    for (int v = 0; v < 8; ++v) {
      mnew[v]  = fmaxf(m[v], cm[v]);
      alpha[v] = __expf(m[v] - mnew[v]);
      rs[v]    = 0.0f;
    }
#pragma unroll
    for (int t = 0; t < 2; ++t)
#pragma unroll
      for (int v = 0; v < 8; ++v) {
        const float p = __expf(sc[t][v] - mnew[v]);
        sc[t][v] = p;
        rs[v] += p;
      }
#pragma unroll
    for (int off = 1; off < 16; off <<= 1)
#pragma unroll
      for (int v = 0; v < 8; ++v) rs[v] += __shfl_xor(rs[v], off, 32);
#pragma unroll
    for (int v = 0; v < 8; ++v) {
      l[v] = l[v] * alpha[v] + rs[v];
      m[v] = mnew[v];
    }
#pragma unroll
    for (int t = 0; t < 4; ++t)
#pragma unroll
      for (int v = 0; v < 8; ++v) o[t][v] *= alpha[v];

    // ---- re-layout P: C-frag -> LDS 16x32 -> A-frag ----
    __syncthreads();
#pragma unroll
    for (int t = 0; t < 2; ++t)
#pragma unroll
      for (int v = 0; v < 8; ++v)
        pw[(v + hi * 8) * 32 + t * 16 + row] = (half_t)sc[t][v];
    __syncthreads();

    v16h pf;
    {
      const half_t* pp = pw + row * 32 + (hi ? 16 : 0);
#pragma unroll
      for (int i = 0; i < 8; ++i) {
        const int kk = 2 * i;
        pf[2 * i]     = pp[kk];
        pf[2 * i + 1] = pp[kk + 1];
      }
    }

    // ---- O += P V : V chunk is 32(keys) x 64(d) -> 4 B fragments ----
#pragma unroll
    for (int t = 0; t < 4; ++t) {
      v16h vf;
      const int dcol = t * 16 + row;
      const half_t* vp = Vp + (size_t)(kc + (hi ? 16 : 0)) * T3E + dcol;
#pragma unroll
      for (int j = 0; j < 8; ++j) {
        vf[2 * j]     = vp[(size_t)(2 * j) * T3E];
        vf[2 * j + 1] = vp[(size_t)(2 * j + 1) * T3E];
      }
      o[t] = __builtin_amdgcn_wmma_f32_16x16x32_f16(false, pf, false, vf,
                                                    (short)0, o[t], false, false);
    }
  }

  // ---- finalize: O / l ----
  const int srow = qt * 16 + hi * 8;
#pragma unroll
  for (int t = 0; t < 4; ++t) {
    const int ocol = h * TD + t * 16 + row;
#pragma unroll
    for (int v = 0; v < 8; ++v) {
      const int s = srow + v;
      out[((size_t)b * TS + s) * TE + ocol] = (half_t)(o[t][v] / l[v]);
    }
  }
}

// ---------------------------------------------------------------------------
// Host orchestration
// ---------------------------------------------------------------------------
extern "C" void kernel_launch(void* const* d_in, const int* in_sizes, int n_in,
                              void* d_out, int out_size, void* d_ws, size_t ws_size,
                              hipStream_t stream) {
  const float* x      = (const float*)d_in[0];
  const float* ln1_w  = (const float*)d_in[1];
  const float* ln1_b  = (const float*)d_in[2];
  const float* qkv_w  = (const float*)d_in[3];
  const float* qkv_b  = (const float*)d_in[4];
  const float* proj_w = (const float*)d_in[5];
  const float* proj_b = (const float*)d_in[6];
  const float* ln2_w  = (const float*)d_in[7];
  const float* ln2_b  = (const float*)d_in[8];
  const float* up_w   = (const float*)d_in[9];
  const float* up_b   = (const float*)d_in[10];
  const float* down_w = (const float*)d_in[11];
  const float* down_b = (const float*)d_in[12];

  // Workspace layout: f32 region first, then f16 region
  float*  x2 = (float*)d_ws;                                    // [M, E] f32
  half_t* hp = (half_t*)(x2 + (size_t)TM * TE);
  half_t* qkv_w16  = hp; hp += (size_t)T3E * TE;
  half_t* proj_w16 = hp; hp += (size_t)TE * TE;
  half_t* up_w16   = hp; hp += (size_t)T4E * TE;
  half_t* down_w16 = hp; hp += (size_t)TE * T4E;
  half_t* hn16     = hp; hp += (size_t)TM * TE;                 // LN output
  half_t* qkv16    = hp; hp += (size_t)TM * T3E;                // qkv
  half_t* attn16   = hp; hp += (size_t)TM * TE;                 // attention out
  half_t* u16      = hp;                                        // MLP hidden

  float* out = (float*)d_out;

  // --- weight casts (f32 -> f16) ---
  cast_kernel<<<(T3E * TE) / 256, 256, 0, stream>>>(qkv_w, qkv_w16);
  cast_kernel<<<(TE * TE) / 256, 256, 0, stream>>>(proj_w, proj_w16);
  cast_kernel<<<(T4E * TE) / 256, 256, 0, stream>>>(up_w, up_w16);
  cast_kernel<<<(TE * T4E) / 256, 256, 0, stream>>>(down_w, down_w16);

  // --- attention sub-block ---
  ln_kernel<<<TM, 256, 0, stream>>>(x, ln1_w, ln1_b, hn16);
  gemm_wmma<<<(TM / 128) * (T3E / 128), 256, 0, stream>>>(
      hn16, qkv_w16, qkv_b, nullptr, nullptr, qkv16, TM, T3E, TE, 0);
  attn_kernel<<<(TB * TH * (TS / 16)) / 4, 128, 0, stream>>>(qkv16, attn16);
  gemm_wmma<<<(TM / 128) * (TE / 128), 256, 0, stream>>>(
      attn16, proj_w16, proj_b, x, x2, nullptr, TM, TE, TE, 0);

  // --- MLP sub-block ---
  ln_kernel<<<TM, 256, 0, stream>>>(x2, ln2_w, ln2_b, hn16);
  gemm_wmma<<<(TM / 128) * (T4E / 128), 256, 0, stream>>>(
      hn16, up_w16, up_b, nullptr, nullptr, u16, TM, T4E, TE, 1);
  gemm_wmma<<<(TM / 128) * (TE / 128), 256, 0, stream>>>(
      u16, down_w16, down_b, x2, out, nullptr, TM, TE, T4E, 0);
}
